// SlotAttention_25194278158455
// MI455X (gfx1250) — matrix operand order repaired and checked by reading, not compiled, not run on hardware
//
#include <hip/hip_runtime.h>
#include <hip/hip_bf16.h>

// ---------------- types for WMMA ----------------
typedef __attribute__((ext_vector_type(16))) __bf16        v16bf;
typedef __attribute__((ext_vector_type(8)))  float         v8f;
typedef __attribute__((ext_vector_type(4)))  unsigned int  v4u;

#define DEV static __device__ __forceinline__

DEV unsigned short f2bf(float f) {
  unsigned u = __float_as_uint(f);
  u += 0x7FFFu + ((u >> 16) & 1u);       // round-to-nearest-even
  return (unsigned short)(u >> 16);
}
DEV float bf2f(unsigned short h) { return __uint_as_float(((unsigned)h) << 16); }
DEV float sigmoidf(float x) { return 1.f / (1.f + expf(-x)); }

// ---------------- problem dims ----------------
static constexpr int Nn   = 100000;
static constexpr int Pp   = 10000;
static constexpr int Ee   = 500000;
static constexpr int DN   = 256;
static constexpr int DP   = 128;
static constexpr int DIN  = 283;   // DN + 27 skip scalars
static constexpr int K1   = 288;   // DIN padded to multiple of 32
static constexpr int DKP  = 112;   // DK=100 padded to multiple of 16
static constexpr int KQ   = 384;   // DP + DN
static constexpr int G3   = 384;   // 3*DP
static constexpr int MLPH = 64;

// =====================================================================
// B pre-pack: fp32 W[rows x cols] -> bf16 WMMA fragments, padded to
// [K x Nc].  Layout: out[((tN*Ksteps + ks)*32 + lane)*8 + j] holds the
// packed bf16 pair (k, k+1) for k = ks*32 + 16*(lane>>4) + 2*j,
// column = tN*16 + (lane&15).  GEMM then reads 2x b128 per lane/K-step.
// =====================================================================
__global__ void pack_b_bf16(const float* __restrict__ W, int rows, int cols,
                            unsigned int* __restrict__ out, int K, int Nc)
{
  int tilesN = Nc >> 4, Ksteps = K >> 5;
  size_t tot = (size_t)tilesN * Ksteps * 256;
  size_t i = blockIdx.x * (size_t)blockDim.x + threadIdx.x;
  if (i >= tot) return;
  int j    = (int)(i & 7);
  int lane = (int)((i >> 3) & 31);
  int rest = (int)(i >> 8);
  int ks = rest % Ksteps;
  int tN = rest / Ksteps;
  int grp = lane >> 4, l15 = lane & 15;
  int kb  = ks * 32 + grp * 16 + 2 * j;
  int col = tN * 16 + l15;
  unsigned lo = (kb     < rows && col < cols) ? f2bf(W[(size_t)kb * cols + col])       : 0u;
  unsigned hi = (kb + 1 < rows && col < cols) ? f2bf(W[(size_t)(kb + 1) * cols + col]) : 0u;
  out[i] = lo | (hi << 16);
}

// =====================================================================
// bf16 WMMA GEMM: C[M,Nc] = A[M,K] * B[K,Nc] + bias (+ReLU)
// A: bf16 row-major (lda = K, 16B-aligned rows), B: pre-packed fragments.
// One wave computes a 64x16 tile (4 M-subtiles x 1 N-tile): 4 independent
// v_wmma_f32_16x16x32_bf16 chains per K-step, B fragment reused 4x.
// M%16==0, Nc%16==0, K%32==0 guaranteed by padding.
// =====================================================================
__global__ __launch_bounds__(256) void gemm_bf16_wmma(
    const unsigned short* __restrict__ A, int lda,
    const unsigned int* __restrict__ Bp,
    const float* __restrict__ bias,
    float* __restrict__ Cf, unsigned short* __restrict__ Cbf,
    int M, int Nc, int K, int relu)
{
  int tilesN  = Nc >> 4;
  int tilesM  = M >> 4;
  int groupsM = (tilesM + 3) >> 2;
  int wave = blockIdx.x * 8 + (threadIdx.x >> 5);
  if (wave >= groupsM * tilesN) return;
  int gM = wave / tilesN;
  int tN = wave - gM * tilesN;

  int lane = threadIdx.x & 31;
  int grp  = lane >> 4;      // 0: lanes 0-15, 1: lanes 16-31
  int l15  = lane & 15;
  int Ksteps = K >> 5;

  const unsigned short* arow[4];
  bool valid[4];
#pragma unroll
  for (int s = 0; s < 4; ++s) {
    int tm   = gM * 4 + s;
    valid[s] = (tm < tilesM);
    int tmc  = valid[s] ? tm : (tilesM - 1);   // clamp: loads stay in-bounds
    arow[s]  = A + (size_t)(tmc * 16 + l15) * lda;
  }
  const unsigned int* bp = Bp + (size_t)tN * Ksteps * 256 + lane * 8;

  v8f acc[4];
#pragma unroll
  for (int s = 0; s < 4; ++s) acc[s] = (v8f){0.f,0.f,0.f,0.f,0.f,0.f,0.f,0.f};

  for (int ks = 0; ks < Ksteps; ++ks) {
    union { v4u q[2]; unsigned int u[8]; v16bf v; } bfr;
    bfr.q[0] = *(const v4u*)(bp + (size_t)ks * 256);
    bfr.q[1] = *(const v4u*)(bp + (size_t)ks * 256 + 4);
    int kk = ks << 5;
#pragma unroll
    for (int s = 0; s < 4; ++s) {
      union { v4u q[2]; unsigned int u[8]; v16bf v; } af;
      af.q[0] = *(const v4u*)(arow[s] + kk + (grp << 4));        // K pairs 16*grp+2*j, j=0..3
      af.q[1] = *(const v4u*)(arow[s] + kk + 8 + (grp << 4));    // j=4..7
      acc[s] = __builtin_amdgcn_wmma_f32_16x16x32_bf16(
          false, af.v, false, bfr.v, (short)0, acc[s], false, false);
    }
  }

  int   col = tN * 16 + l15;
  float bn  = bias ? bias[col] : 0.f;
#pragma unroll
  for (int s = 0; s < 4; ++s) {
    if (!valid[s]) continue;
#pragma unroll
    for (int v = 0; v < 8; ++v) {
      float c = acc[s][v] + bn;
      if (relu) c = fmaxf(c, 0.f);
      int    row = (gM * 4 + s) * 16 + (grp << 3) + v;  // C/D: VGPR v -> M = v + 8*grp
      size_t idx = (size_t)row * Nc + col;
      if (Cf)  Cf[idx]  = c;
      if (Cbf) Cbf[idx] = f2bf(c);
    }
  }
}

// =====================================================================
// Small helper kernels
// =====================================================================
__global__ void convert_bf16(const float* __restrict__ in, unsigned short* __restrict__ out,
                             size_t n)
{
  size_t i = blockIdx.x * (size_t)blockDim.x + threadIdx.x;
  if (i < n) out[i] = f2bf(in[i]);
}

__global__ void pad_bias(const float* __restrict__ in, int n, float* __restrict__ out, int on)
{
  int i = blockIdx.x * blockDim.x + threadIdx.x;
  if (i < on) out[i] = (i < n) ? in[i] : 0.f;
}

__global__ void zero_f32(float* __restrict__ p, size_t n)
{
  size_t i = blockIdx.x * (size_t)blockDim.x + threadIdx.x;
  if (i < n) p[i] = 0.f;
}

// Build x = [node_hidden | 27 skip scalars | zero pad] as bf16 [Nn, K1]
__global__ void build_x_kernel(
    const float* __restrict__ nh, const float* __restrict__ energy,
    const float* __restrict__ eta, const float* __restrict__ phi,
    const float* __restrict__ layer, const float* __restrict__ eta_l,
    const float* __restrict__ phi_l, const float* __restrict__ ene_l,
    const float* __restrict__ trk, unsigned short* __restrict__ out)
{
  size_t i = blockIdx.x * (size_t)blockDim.x + threadIdx.x;
  size_t tot = (size_t)Nn * K1;
  if (i >= tot) return;
  int r = (int)(i / K1), c = (int)(i % K1);
  float v;
  if (c < DN) {
    v = nh[(size_t)r * DN + c];
  } else if (c < DIN) {
    int s = c - DN;
    if      (s == 0) v = (logf(energy[r]) - 4.0f) * 0.5f;       // (log(E)-4)/2
    else if (s == 1) v = eta[r] * (1.0f / 1.5f);                // eta/1.5
    else if (s == 2) v = phi[r];
    else if (s == 3) v = layer[r];
    else if (s < 10) v = eta_l[(size_t)r * 6 + (s - 4)] * (1.0f / 1.5f);
    else if (s < 16) v = phi_l[(size_t)r * 6 + (s - 10)];
    else if (s < 22) v = ene_l[(size_t)r * 6 + (s - 16)];
    else             v = trk[(size_t)r * 5 + (s - 22)];
  } else {
    v = 0.f;
  }
  out[i] = f2bf(v);
}

// Build xq = [particle_hidden | global_rep] bf16 [Pp, KQ]
__global__ void build_xq_kernel(const float* __restrict__ ph, const float* __restrict__ gr,
                                unsigned short* __restrict__ out)
{
  size_t i = blockIdx.x * (size_t)blockDim.x + threadIdx.x;
  size_t tot = (size_t)Pp * KQ;
  if (i >= tot) return;
  int r = (int)(i / KQ), c = (int)(i % KQ);
  float v = (c < DP) ? ph[(size_t)r * DP + c] : gr[(size_t)r * DN + (c - DP)];
  out[i] = f2bf(v);
}

// One wave per edge: att = relu(0.1 * dot(keys[src], q[dst])); scatter att*vals[src]
__global__ __launch_bounds__(256) void edge_kernel(
    const float* __restrict__ keys, const float* __restrict__ q,
    const unsigned short* __restrict__ vals,
    const int* __restrict__ src, const int* __restrict__ dst,
    float* __restrict__ wsacc, int E)
{
  int e = blockIdx.x * 8 + (threadIdx.x >> 5);
  if (e >= E) return;
  int lane = threadIdx.x & 31;
  int s = src[e], d = dst[e];
  const float* kr = keys + (size_t)s * DKP;
  const float* qr = q    + (size_t)d * DKP;
  float acc = kr[lane] * qr[lane]
            + kr[lane + 32] * qr[lane + 32]
            + kr[lane + 64] * qr[lane + 64];
  if (lane < 16) acc += kr[lane + 96] * qr[lane + 96];
#pragma unroll
  for (int o = 16; o; o >>= 1) acc += __shfl_xor(acc, o, 32);
  float att = acc * 0.1f;
  if (att <= 0.f) return;                       // relu: zero contribution
  const unsigned short* vr = vals + (size_t)s * DP;
  float* wr = wsacc + (size_t)d * DP;
#pragma unroll
  for (int t = 0; t < 4; ++t) {
    int j = lane + t * 32;
    atomicAdd(&wr[j], att * bf2f(vr[j]));
  }
}

// Per-particle RMS-norm * sigmoid gate -> bf16
__global__ __launch_bounds__(256) void gate_kernel(
    const float* __restrict__ wsacc, const float* __restrict__ rms_w,
    const float* __restrict__ lin_w, unsigned short* __restrict__ gated, int P_)
{
  int p = blockIdx.x * 8 + (threadIdx.x >> 5);
  if (p >= P_) return;
  int lane = threadIdx.x & 31;
  const float* wr = wsacc + (size_t)p * DP;
  float w[4], ss = 0.f;
#pragma unroll
  for (int t = 0; t < 4; ++t) { w[t] = wr[lane + t * 32]; ss += w[t] * w[t]; }
#pragma unroll
  for (int o = 16; o; o >>= 1) ss += __shfl_xor(ss, o, 32);
  float inv = rsqrtf(ss * (1.f / 128.f) + 1e-6f);
#pragma unroll
  for (int t = 0; t < 4; ++t) {
    int j = lane + t * 32;
    float g = w[t] * inv * rms_w[j] * sigmoidf(w[t] * lin_w[j]);
    gated[(size_t)p * DP + j] = f2bf(g);
  }
}

// GRU combine + LayerNorm -> bf16
__global__ __launch_bounds__(256) void gru_ln_kernel(
    const float* __restrict__ gi, const float* __restrict__ gh,
    const float* __restrict__ h,  const float* __restrict__ ln_g,
    const float* __restrict__ ln_b, unsigned short* __restrict__ lnout, int P_)
{
  int p = blockIdx.x * 8 + (threadIdx.x >> 5);
  if (p >= P_) return;
  int lane = threadIdx.x & 31;
  const float* gir = gi + (size_t)p * G3;
  const float* ghr = gh + (size_t)p * G3;
  const float* hr  = h  + (size_t)p * DP;
  float hn[4], s = 0.f, s2 = 0.f;
#pragma unroll
  for (int t = 0; t < 4; ++t) {
    int j = lane + t * 32;
    float r  = sigmoidf(gir[j]        + ghr[j]);
    float z  = sigmoidf(gir[128 + j]  + ghr[128 + j]);
    float nn = tanhf(gir[256 + j] + r * ghr[256 + j]);
    float hv = (1.f - z) * nn + z * hr[j];
    hn[t] = hv; s += hv; s2 += hv * hv;
  }
#pragma unroll
  for (int o = 16; o; o >>= 1) { s += __shfl_xor(s, o, 32); s2 += __shfl_xor(s2, o, 32); }
  float mu  = s * (1.f / 128.f);
  float var = s2 * (1.f / 128.f) - mu * mu;
  float inv = rsqrtf(var + 1e-5f);
#pragma unroll
  for (int t = 0; t < 4; ++t) {
    int j = lane + t * 32;
    float v = (hn[t] - mu) * inv * ln_g[j] + ln_b[j];
    lnout[(size_t)p * DP + j] = f2bf(v);
  }
}

__global__ void final_add(const float* __restrict__ ph, const float* __restrict__ m2,
                          float* __restrict__ out, size_t n)
{
  size_t i = blockIdx.x * (size_t)blockDim.x + threadIdx.x;
  if (i < n) out[i] = ph[i] + m2[i];
}

// =====================================================================
// Host side
// =====================================================================
static inline int blk1(size_t tot, int t) { return (int)((tot + t - 1) / t); }
static inline int gemm_blocks(int M, int Nc) {
  int waves = (((M >> 4) + 3) >> 2) * (Nc >> 4);
  return (waves + 7) / 8;
}

extern "C" void kernel_launch(void* const* d_in, const int* in_sizes, int n_in,
                              void* d_out, int out_size, void* d_ws, size_t ws_size,
                              hipStream_t stream) {
  (void)in_sizes; (void)n_in; (void)out_size; (void)ws_size;

  const float* node_hidden = (const float*)d_in[0];
  const float* energy      = (const float*)d_in[1];
  const float* eta         = (const float*)d_in[2];
  const float* phi         = (const float*)d_in[3];
  const float* layer       = (const float*)d_in[4];
  const float* eta_l       = (const float*)d_in[5];
  const float* phi_l       = (const float*)d_in[6];
  const float* ene_l       = (const float*)d_in[7];
  const float* trk         = (const float*)d_in[8];
  const float* part_hid    = (const float*)d_in[9];
  const float* global_rep  = (const float*)d_in[10];
  const int*   edge_src    = (const int*)d_in[11];
  const int*   edge_dst    = (const int*)d_in[12];
  const float* W_key = (const float*)d_in[13];
  const float* b_key = (const float*)d_in[14];
  const float* W_val = (const float*)d_in[15];
  const float* b_val = (const float*)d_in[16];
  const float* W_q   = (const float*)d_in[17];
  const float* b_q   = (const float*)d_in[18];
  const float* W_ih  = (const float*)d_in[19];
  const float* b_ih  = (const float*)d_in[20];
  const float* W_hh  = (const float*)d_in[21];
  const float* b_hh  = (const float*)d_in[22];
  const float* ln_g  = (const float*)d_in[23];
  const float* ln_b  = (const float*)d_in[24];
  const float* W1    = (const float*)d_in[25];
  const float* b1    = (const float*)d_in[26];
  const float* W2    = (const float*)d_in[27];
  const float* b2    = (const float*)d_in[28];
  const float* rms_w = (const float*)d_in[29];
  const float* lin_w = (const float*)d_in[30];
  float* out = (float*)d_out;

  // ---- carve workspace ----
  char* base = (char*)d_ws;
  size_t off = 0;
  auto take = [&](size_t bytes) -> void* {
    void* p = base + off;
    off = (off + bytes + 255) & ~(size_t)255;
    return p;
  };
  unsigned short* x_bf   = (unsigned short*)take((size_t)Nn * K1 * 2);
  unsigned short* xq_bf  = (unsigned short*)take((size_t)Pp * KQ * 2);
  unsigned short* h_bf   = (unsigned short*)take((size_t)Pp * DP * 2);
  float*          keys   = (float*)take((size_t)Nn * DKP * 4);
  unsigned short* vals_bf= (unsigned short*)take((size_t)Nn * DP * 2);
  float*          qm     = (float*)take((size_t)Pp * DKP * 4);
  float*          wsacc  = (float*)take((size_t)Pp * DP * 4);
  unsigned short* gated  = (unsigned short*)take((size_t)Pp * DP * 2);
  float*          gi     = (float*)take((size_t)Pp * G3 * 4);
  float*          gh     = (float*)take((size_t)Pp * G3 * 4);
  unsigned short* lnb16  = (unsigned short*)take((size_t)Pp * DP * 2);
  unsigned short* m1b    = (unsigned short*)take((size_t)Pp * MLPH * 2);
  float*          m2     = (float*)take((size_t)Pp * DP * 4);
  unsigned int*   Wk_p   = (unsigned int*)take((size_t)K1 * DKP * 2);
  unsigned int*   Wv_p   = (unsigned int*)take((size_t)K1 * DP * 2);
  unsigned int*   Wq_p   = (unsigned int*)take((size_t)KQ * DKP * 2);
  unsigned int*   Wih_p  = (unsigned int*)take((size_t)DP * G3 * 2);
  unsigned int*   Whh_p  = (unsigned int*)take((size_t)DP * G3 * 2);
  unsigned int*   W1_p   = (unsigned int*)take((size_t)DP * MLPH * 2);
  unsigned int*   W2_p   = (unsigned int*)take((size_t)MLPH * DP * 2);
  float*          bk_pad = (float*)take((size_t)DKP * 4);
  float*          bq_pad = (float*)take((size_t)DKP * 4);

  const int T = 256;

  // ---- weight conversion + fragment packing ----
  pack_b_bf16<<<blk1((size_t)(DKP/16)*(K1/32)*256, T), T, 0, stream>>>(W_key, DIN, 100, Wk_p, K1, DKP);
  pack_b_bf16<<<blk1((size_t)(DP /16)*(K1/32)*256, T), T, 0, stream>>>(W_val, DIN, DP,  Wv_p, K1, DP);
  pack_b_bf16<<<blk1((size_t)(DKP/16)*(KQ/32)*256, T), T, 0, stream>>>(W_q,   KQ,  100, Wq_p, KQ, DKP);
  pack_b_bf16<<<blk1((size_t)(G3 /16)*(DP/32)*256, T), T, 0, stream>>>(W_ih,  DP,  G3,  Wih_p, DP, G3);
  pack_b_bf16<<<blk1((size_t)(G3 /16)*(DP/32)*256, T), T, 0, stream>>>(W_hh,  DP,  G3,  Whh_p, DP, G3);
  pack_b_bf16<<<blk1((size_t)(MLPH/16)*(DP/32)*256, T), T, 0, stream>>>(W1,   DP,  MLPH, W1_p, DP, MLPH);
  pack_b_bf16<<<blk1((size_t)(DP /16)*(MLPH/32)*256, T), T, 0, stream>>>(W2,  MLPH, DP,  W2_p, MLPH, DP);
  pad_bias<<<1, DKP, 0, stream>>>(b_key, 100, bk_pad, DKP);
  pad_bias<<<1, DKP, 0, stream>>>(b_q,   100, bq_pad, DKP);

  // ---- build activations ----
  build_x_kernel<<<blk1((size_t)Nn * K1, T), T, 0, stream>>>(
      node_hidden, energy, eta, phi, layer, eta_l, phi_l, ene_l, trk, x_bf);
  build_xq_kernel<<<blk1((size_t)Pp * KQ, T), T, 0, stream>>>(part_hid, global_rep, xq_bf);
  convert_bf16<<<blk1((size_t)Pp * DP, T), T, 0, stream>>>(part_hid, h_bf, (size_t)Pp * DP);

  // ---- keys / vals / q GEMMs ----
  gemm_bf16_wmma<<<gemm_blocks(Nn, DKP), T, 0, stream>>>(
      x_bf, K1, Wk_p, bk_pad, keys, nullptr, Nn, DKP, K1, 0);
  gemm_bf16_wmma<<<gemm_blocks(Nn, DP), T, 0, stream>>>(
      x_bf, K1, Wv_p, b_val, nullptr, vals_bf, Nn, DP, K1, 0);
  gemm_bf16_wmma<<<gemm_blocks(Pp, DKP), T, 0, stream>>>(
      xq_bf, KQ, Wq_p, bq_pad, qm, nullptr, Pp, DKP, KQ, 0);

  // ---- edge attention + segment sum ----
  zero_f32<<<blk1((size_t)Pp * DP, T), T, 0, stream>>>(wsacc, (size_t)Pp * DP);
  edge_kernel<<<(Ee + 7) / 8, T, 0, stream>>>(keys, qm, vals_bf, edge_src, edge_dst, wsacc, Ee);

  // ---- gate ----
  gate_kernel<<<(Pp + 7) / 8, T, 0, stream>>>(wsacc, rms_w, lin_w, gated, Pp);

  // ---- GRU GEMMs ----
  gemm_bf16_wmma<<<gemm_blocks(Pp, G3), T, 0, stream>>>(
      gated, DP, Wih_p, b_ih, gi, nullptr, Pp, G3, DP, 0);
  gemm_bf16_wmma<<<gemm_blocks(Pp, G3), T, 0, stream>>>(
      h_bf, DP, Whh_p, b_hh, gh, nullptr, Pp, G3, DP, 0);

  // ---- GRU combine + LayerNorm ----
  gru_ln_kernel<<<(Pp + 7) / 8, T, 0, stream>>>(gi, gh, part_hid, ln_g, ln_b, lnb16, Pp);

  // ---- MLP ----
  gemm_bf16_wmma<<<gemm_blocks(Pp, MLPH), T, 0, stream>>>(
      lnb16, DP, W1_p, b1, nullptr, m1b, Pp, MLPH, DP, 1);
  gemm_bf16_wmma<<<gemm_blocks(Pp, DP), T, 0, stream>>>(
      m1b, MLPH, W2_p, b2, m2, nullptr, Pp, DP, MLPH, 0);

  // ---- residual ----
  final_add<<<blk1((size_t)Pp * DP, T), T, 0, stream>>>(part_hid, m2, out, (size_t)Pp * DP);
}
